// SwInception_85718957293895
// MI455X (gfx1250) — matrix operand order repaired and checked by reading, not compiled, not run on hardware
//
#include <hip/hip_runtime.h>
#include <hip/hip_bf16.h>
#include <math.h>

// ---------------------------------------------------------------------------
// CDNA5 (gfx1250) wave32 WMMA bf16 implementation of SwInception block.
// ---------------------------------------------------------------------------

typedef __bf16 v16bf __attribute__((ext_vector_type(16)));
typedef float  v8f   __attribute__((ext_vector_type(8)));
typedef unsigned short ushort8v __attribute__((ext_vector_type(8)));
typedef int    v4i   __attribute__((ext_vector_type(4)));

union AFrag { v16bf v; ushort8v h[2]; unsigned short u[16]; };

// Async global->LDS copy path (ASYNCcnt) if the toolchain exposes it.
#if defined(__has_builtin)
#  if __has_builtin(__builtin_amdgcn_global_load_async_to_lds_b128) && \
      __has_builtin(__builtin_amdgcn_s_wait_asynccnt)
#    define USE_ASYNC_LDS 1
#  endif
#endif
#ifndef USE_ASYNC_LDS
#  define USE_ASYNC_LDS 0
#endif

#if USE_ASYNC_LDS
// Builtin signature (from compiler diagnostic): 128-bit payload pointers:
//   (v4i addrspace(1)* gsrc, v4i addrspace(3)* lds, imm offset, imm cpol)
__device__ __forceinline__ void async_g2l_b128(const void* g, void* l) {
  __builtin_amdgcn_global_load_async_to_lds_b128(
      (__attribute__((address_space(1))) v4i*)g,
      (__attribute__((address_space(3))) v4i*)l, 0, 0);
}
#endif

__device__ __forceinline__ unsigned short f2bf(float f) {
  __bf16 b = (__bf16)f;          // native f32->bf16 RNE convert on gfx1250
  unsigned short h;
  __builtin_memcpy(&h, &b, 2);
  return h;
}
__device__ __forceinline__ float bf2f(unsigned short h) {
  union { unsigned u; float f; } c; c.u = ((unsigned)h) << 16;
  return c.f;
}

__device__ __forceinline__ v8f wmma_bf16(v16bf a, v16bf b, v8f c) {
  // D = A(16x32) * B(32x16) + C  in fp32, wave32
  return __builtin_amdgcn_wmma_f32_16x16x32_bf16(false, a, false, b, (short)0, c,
                                                 false, false);
}

// A fragment from row-major [M x ld] bf16 storage (contiguous k order, as in the B loaders):
// lane m = lane&15, half = lane>>4 ; element e -> k = k0 + 16*half + e
__device__ __forceinline__ v16bf load_a_rm(const unsigned short* base, int ld,
                                           int m0, int k0) {
  int lane = threadIdx.x & 31, m = lane & 15, hf = lane >> 4;
  const unsigned short* p = base + (size_t)(m0 + m) * ld + k0 + 16 * hf;
  AFrag f;
  f.h[0] = *(const ushort8v*)p;         // k0+16h .. +7
  f.h[1] = *(const ushort8v*)(p + 8);   // k0+16h+8 .. +15
  return f.v;
}
// B fragment where memory is [N x ld] row-major (B[k][n] = W[n][k]):
// lane n = lane&15, half = lane>>4 ; element e -> k = k0 + 16*half + e
__device__ __forceinline__ v16bf load_b_nk(const unsigned short* W, int ld,
                                           int n0, int k0) {
  int lane = threadIdx.x & 31, n = lane & 15, hf = lane >> 4;
  const unsigned short* p = W + (size_t)(n0 + n) * ld + k0 + 16 * hf;
  AFrag f;
  f.h[0] = *(const ushort8v*)p;
  f.h[1] = *(const ushort8v*)(p + 8);
  return f.v;
}
// B fragment where memory is [K x ld] (B[k][n] = V[k*ld + n]) -- strided gather
__device__ __forceinline__ v16bf load_b_kn(const unsigned short* V, int ld,
                                           int k0, int n0) {
  int lane = threadIdx.x & 31, n = lane & 15, hf = lane >> 4;
  AFrag f;
#pragma unroll
  for (int e = 0; e < 16; ++e)
    f.u[e] = V[(size_t)(k0 + 16 * hf + e) * ld + n0 + n];
  return f.v;
}

// ---------------------------------------------------------------------------
// weight / bias preparation
// ---------------------------------------------------------------------------
__global__ void k_cvt_pad(const float* __restrict__ src, unsigned short* __restrict__ dst,
                          int N, int K, int Npad) {
  size_t id = (size_t)blockIdx.x * 256 + threadIdx.x;
  if (id >= (size_t)Npad * K) return;
  int n = (int)(id / K);
  float v = (n < N) ? src[id] : 0.f;
  dst[id] = f2bf(v);
}

// pack conv weights [cout][24][3][3][3] fp32 -> [27][coutPad][32] bf16 (cin padded)
__global__ void k_prep_convw(const float* __restrict__ w, unsigned short* __restrict__ dst,
                             int cout, int coutPad) {
  size_t id = (size_t)blockIdx.x * 256 + threadIdx.x;
  size_t tot = (size_t)27 * coutPad * 32;
  if (id >= tot) return;
  int o  = (int)(id / (coutPad * 32));
  int r  = (int)(id % (coutPad * 32));
  int co = r >> 5, ci = r & 31;
  float v = 0.f;
  if (co < cout && ci < 24) v = w[((size_t)co * 24 + ci) * 27 + o];
  dst[id] = f2bf(v);
}

// bias6[h][n][m] = rpb[RPI[n,m]*6 + h], RPI recomputed on device
__global__ void k_prep_bias(const float* __restrict__ rpb, float* __restrict__ bias6) {
  int id = blockIdx.x * 256 + threadIdx.x;  // 6*64*64
  int h = id >> 12;
  int n = (id >> 6) & 63, m = id & 63;
  int ni = n >> 4, nj = (n >> 2) & 3, nk = n & 3;
  int mi = m >> 4, mj = (m >> 2) & 3, mk = m & 3;
  int idx = ((ni - mi + 3) * 7 + (nj - mj + 3)) * 7 + (nk - mk + 3);
  bias6[id] = rpb[idx * 6 + h];
}

// ---------------------------------------------------------------------------
// fused windowed attention: LN1 -> QKV -> (QK^T+bias, softmax, PV) -> proj + res
// one block = one 4x4x4 window (64 tokens), 256 threads = 8 wave32 waves
// ---------------------------------------------------------------------------
__global__ void k_attn(const float* __restrict__ x,
                       const float* __restrict__ g1, const float* __restrict__ be1,
                       const unsigned short* __restrict__ wqkv,
                       const float* __restrict__ qkvb,
                       const float* __restrict__ bias6,
                       const unsigned short* __restrict__ wproj,
                       const float* __restrict__ projb,
                       float* __restrict__ xattn) {
  extern __shared__ char smem[];
  unsigned short* xln  = (unsigned short*)smem;             // 64*192 bf16 (24.0 KB)
  unsigned short* qkvs = xln + 64 * 192;                    // 64*576 bf16 (72.0 KB)
  float*          sc   = (float*)(qkvs + 64 * 576);         // 64*64 f32  (16.0 KB)
  unsigned short* scp  = (unsigned short*)sc;               // probs overlay (bf16)
  unsigned short* obuf = (unsigned short*)(sc + 64 * 64);   // 64*192 bf16 (24.0 KB)

  int wid = blockIdx.x;
  int b  = wid >> 9;
  int w3 = wid & 511;
  int wz = w3 >> 6, wy = (w3 >> 3) & 7, wx = w3 & 7;
  int tid = threadIdx.x, lane = tid & 31, wave = tid >> 5;

  // ---- phase 1: LayerNorm (4 threads per token, 48 ch each) ----
  {
    int lt = tid >> 2, q = tid & 3;
    int tz = lt >> 4, ty = (lt >> 2) & 3, tx = lt & 3;
    int l = ((wz * 4 + tz) * 32 + (wy * 4 + ty)) * 32 + (wx * 4 + tx);
    const float* px = x + ((size_t)b * 32768 + l) * 192 + q * 48;
    float xv[48]; float s = 0.f, s2 = 0.f;
#pragma unroll
    for (int c = 0; c < 48; ++c) { float v = px[c]; xv[c] = v; s += v; s2 += v * v; }
    s  += __shfl_xor(s, 1, 32);  s  += __shfl_xor(s, 2, 32);
    s2 += __shfl_xor(s2, 1, 32); s2 += __shfl_xor(s2, 2, 32);
    float mu   = s * (1.f / 192.f);
    float var  = s2 * (1.f / 192.f) - mu * mu;
    float rstd = rsqrtf(var + 1e-5f);
#pragma unroll
    for (int c = 0; c < 48; ++c) {
      int ch = q * 48 + c;
      xln[lt * 192 + ch] = f2bf((xv[c] - mu) * rstd * g1[ch] + be1[ch]);
    }
  }
  __syncthreads();

  // ---- phase 2: QKV GEMM [64x192]x[192x576], q scaled ----
  for (int tile = wave; tile < 4 * 36; tile += 8) {
    int m0 = (tile / 36) * 16, n0 = (tile % 36) * 16;
    v8f acc;
#pragma unroll
    for (int r = 0; r < 8; ++r) acc[r] = 0.f;
#pragma unroll
    for (int ks = 0; ks < 6; ++ks) {
      v16bf a  = load_a_rm(xln, 192, m0, ks * 32);
      v16bf bb = load_b_nk(wqkv, 192, n0, ks * 32);
      acc = wmma_bf16(a, bb, acc);
    }
    int hf = lane >> 4, nn = lane & 15;
    int col = n0 + nn;
    float bias = qkvb[col];
    float scl  = (col < 192) ? 0.17677669529663687f : 1.f;  // 1/sqrt(32) on q
#pragma unroll
    for (int r = 0; r < 8; ++r)
      qkvs[(m0 + r + 8 * hf) * 576 + col] = f2bf((acc[r] + bias) * scl);
  }
  __syncthreads();

  // ---- phase 3: per-head QK^T + bias, softmax, PV ----
  for (int h = 0; h < 6; ++h) {
    // scores (16 tiles, 2 per wave), C initialized with rel-pos bias
    for (int tile = wave * 2; tile < wave * 2 + 2; ++tile) {
      int m0 = (tile >> 2) * 16, n0 = (tile & 3) * 16;
      int hf = lane >> 4, nn = lane & 15;
      v8f acc;
#pragma unroll
      for (int r = 0; r < 8; ++r)
        acc[r] = bias6[h * 4096 + (m0 + r + 8 * hf) * 64 + n0 + nn];
      v16bf a  = load_a_rm(qkvs, 576, m0, h * 32);                 // q
      v16bf bb = load_b_nk(qkvs + 192 + h * 32, 576, n0, 0);       // k^T
      acc = wmma_bf16(a, bb, acc);
#pragma unroll
      for (int r = 0; r < 8; ++r)
        sc[(m0 + r + 8 * hf) * 64 + n0 + nn] = acc[r];
    }
    __syncthreads();
    // softmax: 4 threads per row, 16 cols each
    {
      int row = tid >> 2, q = tid & 3;
      float v[16]; float mx = -1e30f;
#pragma unroll
      for (int c = 0; c < 16; ++c) { v[c] = sc[row * 64 + q * 16 + c]; mx = fmaxf(mx, v[c]); }
      mx = fmaxf(mx, __shfl_xor(mx, 1, 32));
      mx = fmaxf(mx, __shfl_xor(mx, 2, 32));
      float s = 0.f;
#pragma unroll
      for (int c = 0; c < 16; ++c) { v[c] = expf(v[c] - mx); s += v[c]; }
      s += __shfl_xor(s, 1, 32); s += __shfl_xor(s, 2, 32);
      float inv = 1.f / s;
      __syncthreads();  // all fp32 reads done before bf16 overlay writes
#pragma unroll
      for (int c = 0; c < 16; ++c) scp[row * 64 + q * 16 + c] = f2bf(v[c] * inv);
    }
    __syncthreads();
    // out_h = probs[64x64] @ v[64x32]  (8 tiles, 1 per wave)
    {
      int m0 = (wave >> 1) * 16, n0 = (wave & 1) * 16;
      v8f acc;
#pragma unroll
      for (int r = 0; r < 8; ++r) acc[r] = 0.f;
#pragma unroll
      for (int ks = 0; ks < 2; ++ks) {
        v16bf a  = load_a_rm(scp, 64, m0, ks * 32);
        v16bf bb = load_b_kn(qkvs + 384 + h * 32, 576, ks * 32, n0);  // v
        acc = wmma_bf16(a, bb, acc);
      }
      int hf = lane >> 4, nn = lane & 15;
#pragma unroll
      for (int r = 0; r < 8; ++r)
        obuf[(m0 + r + 8 * hf) * 192 + h * 32 + n0 + nn] = f2bf(acc[r]);
    }
    __syncthreads();
  }

  // ---- phase 4: proj GEMM + bias + residual, scatter back to [B,L,C] ----
  for (int tile = wave; tile < 48; tile += 8) {
    int m0 = (tile / 12) * 16, n0 = (tile % 12) * 16;
    v8f acc;
#pragma unroll
    for (int r = 0; r < 8; ++r) acc[r] = 0.f;
#pragma unroll
    for (int ks = 0; ks < 6; ++ks) {
      v16bf a  = load_a_rm(obuf, 192, m0, ks * 32);
      v16bf bb = load_b_nk(wproj, 192, n0, ks * 32);
      acc = wmma_bf16(a, bb, acc);
    }
    int hf = lane >> 4, nn = lane & 15;
    int col = n0 + nn;
    float pb = projb[col];
#pragma unroll
    for (int r = 0; r < 8; ++r) {
      int lt = m0 + r + 8 * hf;
      int tz = lt >> 4, ty = (lt >> 2) & 3, tx = lt & 3;
      int l = ((wz * 4 + tz) * 32 + (wy * 4 + ty)) * 32 + (wx * 4 + tx);
      size_t idx = ((size_t)b * 32768 + l) * 192 + col;
      xattn[idx] = acc[r] + pb + x[idx];
    }
  }
}

// ---------------------------------------------------------------------------
// standalone LayerNorm (token-major), fp32 in -> bf16 out
// ---------------------------------------------------------------------------
__global__ void k_ln(const float* __restrict__ in, const float* __restrict__ g,
                     const float* __restrict__ be, unsigned short* __restrict__ out) {
  int tid = threadIdx.x;
  size_t tk = (size_t)blockIdx.x * 64 + (tid >> 2);
  int q = tid & 3;
  const float* p = in + tk * 192 + q * 48;
  float xv[48]; float s = 0.f, s2 = 0.f;
#pragma unroll
  for (int c = 0; c < 48; ++c) { float v = p[c]; xv[c] = v; s += v; s2 += v * v; }
  s  += __shfl_xor(s, 1, 32);  s  += __shfl_xor(s, 2, 32);
  s2 += __shfl_xor(s2, 1, 32); s2 += __shfl_xor(s2, 2, 32);
  float mu = s * (1.f / 192.f);
  float rstd = rsqrtf(s2 * (1.f / 192.f) - mu * mu + 1e-5f);
#pragma unroll
  for (int c = 0; c < 48; ++c) {
    int ch = q * 48 + c;
    out[tk * 192 + ch] = f2bf((xv[c] - mu) * rstd * g[ch] + be[ch]);
  }
}

// ---------------------------------------------------------------------------
// generic token-major 1x1 GEMM: out[t,n] = sum_k A[t,k]*W[n,k] + bias[n]
// ---------------------------------------------------------------------------
__global__ void k_gemm(const unsigned short* __restrict__ A,
                       const unsigned short* __restrict__ W,
                       int K, int Ntiles,
                       const float* __restrict__ bias,
                       float* __restrict__ out, int ldo, int nOut) {
  int tid = threadIdx.x, lane = tid & 31, wave = tid >> 5;
  size_t tokBase = (size_t)blockIdx.x * 64;
  const unsigned short* Ab = A + tokBase * K;
  int ksteps = K >> 5;
  for (int tile = wave; tile < 4 * Ntiles; tile += 8) {
    int m0 = (tile / Ntiles) * 16, n0 = (tile % Ntiles) * 16;
    v8f acc;
#pragma unroll
    for (int r = 0; r < 8; ++r) acc[r] = 0.f;
    for (int ks = 0; ks < ksteps; ++ks) {
      v16bf a  = load_a_rm(Ab, K, m0, ks * 32);
      v16bf bb = load_b_nk(W, K, n0, ks * 32);
      acc = wmma_bf16(a, bb, acc);
    }
    int hf = lane >> 4, nn = lane & 15, col = n0 + nn;
    if (col < nOut) {
      float bv = bias[col];
#pragma unroll
      for (int r = 0; r < 8; ++r)
        out[(tokBase + m0 + r + 8 * hf) * ldo + col] = acc[r] + bv;
    }
  }
}

// ---------------------------------------------------------------------------
// 3x3x3 conv (cin=24 padded to K=32 per tap) via WMMA over 27 K-chunks.
// A-tile staged into LDS through the gfx1250 async global->LDS engine
// (ASYNCcnt) when available; OOB lanes zero their LDS slots via DS stores.
// ---------------------------------------------------------------------------
__global__ void k_conv3(const unsigned short* __restrict__ tin,   // [65536][24] bf16
                        const unsigned short* __restrict__ wB,    // [27][coutPad][32] bf16
                        const float* __restrict__ bias,
                        float* __restrict__ out, int ldo, int cout, int coutPad) {
  __shared__ unsigned short As[64 * 32];
  int tid = threadIdx.x, lane = tid & 31, wave = tid >> 5;
  size_t tokBase = (size_t)blockIdx.x * 64;
  int lt = tid >> 2, j = tid & 3;
  int tk = (int)tokBase + lt;
  int bb = tk >> 15, pos = tk & 32767;
  int z = pos >> 10, y = (pos >> 5) & 31, xx = pos & 31;
  int Nt = coutPad >> 4;
  int mine = (4 * Nt) >> 3;  // tiles per wave: 6 (cout=192) or 1 (cout=24)
  v8f acc[6];
  for (int i = 0; i < mine; ++i) {
#pragma unroll
    for (int r = 0; r < 8; ++r) acc[i][r] = 0.f;
  }
  for (int o = 0; o < 27; ++o) {
    int dz = o / 9 - 1, dy = (o / 3) % 3 - 1, dx = o % 3 - 1;
    int nz = z + dz, ny = y + dy, nx = xx + dx;
    bool valid = (j < 3) && nz >= 0 && nz < 32 && ny >= 0 && ny < 32 &&
                 nx >= 0 && nx < 32;
    size_t tn = (size_t)bb * 32768 + ((nz * 32 + ny) * 32 + nx);
#if USE_ASYNC_LDS
    if (valid) {
      async_g2l_b128(tin + tn * 24 + j * 8, As + lt * 32 + j * 8);
    } else {
      ushort8v zv;
#pragma unroll
      for (int e = 0; e < 8; ++e) zv[e] = 0;
      *(ushort8v*)(As + lt * 32 + j * 8) = zv;   // disjoint from async dests
    }
    __builtin_amdgcn_s_wait_asynccnt(0);
    __syncthreads();
#else
    ushort8v v;
#pragma unroll
    for (int e = 0; e < 8; ++e) v[e] = 0;
    if (valid) v = *(const ushort8v*)(tin + tn * 24 + j * 8);
    *(ushort8v*)(As + lt * 32 + j * 8) = v;
    __syncthreads();
#endif
    const unsigned short* Wo = wB + (size_t)o * coutPad * 32;
    for (int i = 0; i < mine; ++i) {
      int tile = wave * mine + i;
      int m0 = (tile / Nt) * 16, n0 = (tile % Nt) * 16;
      v16bf a  = load_a_rm(As, 32, m0, 0);
      v16bf bw = load_b_nk(Wo, 32, n0, 0);
      acc[i] = wmma_bf16(a, bw, acc[i]);
    }
    __syncthreads();
  }
  int hf = lane >> 4, nn = lane & 15;
  for (int i = 0; i < mine; ++i) {
    int tile = wave * mine + i;
    int m0 = (tile / Nt) * 16, n0 = (tile % Nt) * 16;
    int col = n0 + nn;
    if (col < cout) {
      float bv = bias[col];
#pragma unroll
      for (int r = 0; r < 8; ++r)
        out[(tokBase + m0 + r + 8 * hf) * ldo + col] = acc[i][r] + bv;
    }
  }
}

// ---------------------------------------------------------------------------
// BatchNorm stats (deterministic 2-stage tree), then BN+GELU(exact) -> bf16
// ---------------------------------------------------------------------------
__global__ void k_stat1(const float* __restrict__ conv, int ldc,
                        float* __restrict__ psum, float* __restrict__ psq) {
  __shared__ float ss[256], sq[256];
  int c = blockIdx.x, j = blockIdx.y;
  int tid = threadIdx.x;
  size_t t0 = (size_t)j * 512 + tid * 2;
  float a = conv[t0 * ldc + c], b = conv[(t0 + 1) * ldc + c];
  ss[tid] = a + b; sq[tid] = a * a + b * b;
  __syncthreads();
  for (int s = 128; s > 0; s >>= 1) {
    if (tid < s) { ss[tid] += ss[tid + s]; sq[tid] += sq[tid + s]; }
    __syncthreads();
  }
  if (tid == 0) { psum[(size_t)c * 128 + j] = ss[0]; psq[(size_t)c * 128 + j] = sq[0]; }
}

__global__ void k_stat2(const float* __restrict__ psum, const float* __restrict__ psq,
                        float* __restrict__ stats) {
  __shared__ float ss[128], sq[128];
  int c = blockIdx.x, tid = threadIdx.x;
  ss[tid] = psum[(size_t)c * 128 + tid];
  sq[tid] = psq[(size_t)c * 128 + tid];
  __syncthreads();
  for (int s = 64; s > 0; s >>= 1) {
    if (tid < s) { ss[tid] += ss[tid + s]; sq[tid] += sq[tid + s]; }
    __syncthreads();
  }
  if (tid == 0) {
    float mu  = ss[0] * (1.f / 65536.f);
    float var = sq[0] * (1.f / 65536.f) - mu * mu;
    stats[c * 2]     = mu;
    stats[c * 2 + 1] = rsqrtf(var + 1e-3f);
  }
}

__global__ void k_bngelu(const float* __restrict__ conv, int C,
                         const float* __restrict__ stats,
                         const float* __restrict__ g, const float* __restrict__ be,
                         unsigned short* __restrict__ dst, int dstStride, int dstOff) {
  size_t id = (size_t)blockIdx.x * 256 + threadIdx.x;
  size_t tk = id / C; int c = (int)(id % C);
  float v = conv[id];
  float y = (v - stats[c * 2]) * stats[c * 2 + 1] * g[c] + be[c];
  float ge = 0.5f * y * (1.f + erff(y * 0.70710678118654752f));
  dst[tk * dstStride + dstOff + c] = f2bf(ge);
}

// ---------------------------------------------------------------------------
// 3x3x3 avg pool (zero padded, /27) on token-major bf16
// ---------------------------------------------------------------------------
__global__ void k_pool(const unsigned short* __restrict__ h,
                       unsigned short* __restrict__ pooled) {
  int tid = threadIdx.x;
  int tk = blockIdx.x * 64 + (tid >> 2), q = tid & 3;
  int b = tk >> 15, pos = tk & 32767;
  int z = pos >> 10, y = (pos >> 5) & 31, x = pos & 31;
  float acc[48];
#pragma unroll
  for (int c = 0; c < 48; ++c) acc[c] = 0.f;
  for (int dz = -1; dz <= 1; ++dz)
    for (int dy = -1; dy <= 1; ++dy)
      for (int dx = -1; dx <= 1; ++dx) {
        int nz = z + dz, ny = y + dy, nx = x + dx;
        if (nz < 0 || nz >= 32 || ny < 0 || ny >= 32 || nx < 0 || nx >= 32) continue;
        const unsigned short* p =
            h + ((size_t)b * 32768 + ((nz * 32 + ny) * 32 + nx)) * 192 + q * 48;
#pragma unroll
        for (int c = 0; c < 48; ++c) acc[c] += bf2f(p[c]);
      }
  unsigned short* o = pooled + (size_t)tk * 192 + q * 48;
#pragma unroll
  for (int c = 0; c < 48; ++c) o[c] = f2bf(acc[c] * (1.f / 27.f));
}

// ---------------------------------------------------------------------------
// final fc: out = cat[65536,768] @ fc_w^T + fc_b + xattn
// ---------------------------------------------------------------------------
__global__ void k_fc(const unsigned short* __restrict__ catb,
                     const unsigned short* __restrict__ wfc,
                     const float* __restrict__ fcb,
                     const float* __restrict__ xr,
                     float* __restrict__ out) {
  int tid = threadIdx.x, lane = tid & 31, wave = tid >> 5;
  size_t tokBase = (size_t)blockIdx.x * 64;
  const unsigned short* Ab = catb + tokBase * 768;
  for (int tile = wave; tile < 48; tile += 8) {
    int m0 = (tile / 12) * 16, n0 = (tile % 12) * 16;
    v8f acc;
#pragma unroll
    for (int r = 0; r < 8; ++r) acc[r] = 0.f;
    for (int ks = 0; ks < 24; ++ks) {
      v16bf a  = load_a_rm(Ab, 768, m0, ks * 32);
      v16bf bb = load_b_nk(wfc, 768, n0, ks * 32);
      acc = wmma_bf16(a, bb, acc);
    }
    int hf = lane >> 4, nn = lane & 15, col = n0 + nn;
    float bv = fcb[col];
#pragma unroll
    for (int r = 0; r < 8; ++r) {
      size_t idx = (tokBase + m0 + r + 8 * hf) * 192 + col;
      out[idx] = acc[r] + bv + xr[idx];
    }
  }
}

// ---------------------------------------------------------------------------
#define CEILDIV(a, b) (((a) + (b) - 1) / (b))

extern "C" void kernel_launch(void* const* d_in, const int* in_sizes, int n_in,
                              void* d_out, int out_size, void* d_ws, size_t ws_size,
                              hipStream_t stream) {
  (void)in_sizes; (void)n_in; (void)out_size; (void)ws_size;
  const float* x      = (const float*)d_in[0];
  const float* g1     = (const float*)d_in[1];
  const float* beta1  = (const float*)d_in[2];
  const float* qkv_w  = (const float*)d_in[3];
  const float* qkv_b  = (const float*)d_in[4];
  const float* rpb    = (const float*)d_in[5];
  const float* proj_w = (const float*)d_in[6];
  const float* proj_b = (const float*)d_in[7];
  const float* g2     = (const float*)d_in[8];
  const float* beta2  = (const float*)d_in[9];
  const float* fc_w   = (const float*)d_in[10];
  const float* fc_b   = (const float*)d_in[11];
  const float* b1_w  = (const float*)d_in[12];
  const float* b1_b  = (const float*)d_in[13];
  const float* b1_g  = (const float*)d_in[14];
  const float* b1_be = (const float*)d_in[15];
  const float* b3a_w = (const float*)d_in[16];
  const float* b3a_b = (const float*)d_in[17];
  const float* b3a_g = (const float*)d_in[18];
  const float* b3a_be= (const float*)d_in[19];
  const float* b3b_w = (const float*)d_in[20];
  const float* b3b_b = (const float*)d_in[21];
  const float* b3b_g = (const float*)d_in[22];
  const float* b3b_be= (const float*)d_in[23];
  const float* b5a_w = (const float*)d_in[24];
  const float* b5a_b = (const float*)d_in[25];
  const float* b5a_g = (const float*)d_in[26];
  const float* b5a_be= (const float*)d_in[27];
  const float* b5b_w = (const float*)d_in[28];
  const float* b5b_b = (const float*)d_in[29];
  const float* b5b_g = (const float*)d_in[30];
  const float* b5b_be= (const float*)d_in[31];
  const float* b5c_w = (const float*)d_in[32];
  const float* b5c_b = (const float*)d_in[33];
  const float* b5c_g = (const float*)d_in[34];
  const float* b5c_be= (const float*)d_in[35];
  const float* bp_w  = (const float*)d_in[36];
  const float* bp_b  = (const float*)d_in[37];
  const float* bp_g  = (const float*)d_in[38];
  const float* bp_be = (const float*)d_in[39];
  float* out = (float*)d_out;

  unsigned char* base = (unsigned char*)d_ws;
  size_t off = 0;
  auto alloc = [&](size_t bytes) -> void* {
    void* p = base + off;
    off += (bytes + 255) & ~(size_t)255;
    return p;
  };
  const size_t TOK = 65536;  // B * L
  unsigned short* wqkv  = (unsigned short*)alloc(576 * 192 * 2);
  unsigned short* wproj = (unsigned short*)alloc(192 * 192 * 2);
  unsigned short* wfc   = (unsigned short*)alloc(192 * 768 * 2);
  unsigned short* wb1   = (unsigned short*)alloc(192 * 192 * 2);
  unsigned short* wb3a  = (unsigned short*)alloc(32 * 192 * 2);
  unsigned short* wb5a  = (unsigned short*)alloc(32 * 192 * 2);
  unsigned short* wbp   = (unsigned short*)alloc(192 * 192 * 2);
  unsigned short* wb3b  = (unsigned short*)alloc(27 * 192 * 32 * 2);
  unsigned short* wb5b  = (unsigned short*)alloc(27 * 32 * 32 * 2);
  unsigned short* wb5c  = (unsigned short*)alloc(27 * 192 * 32 * 2);
  float*          bias6 = (float*)alloc(6 * 64 * 64 * 4);
  float*          xattn = (float*)alloc(TOK * 192 * 4);
  unsigned short* hbuf  = (unsigned short*)alloc(TOK * 192 * 2);
  unsigned short* catb  = (unsigned short*)alloc(TOK * 768 * 2);
  unsigned short* pooled= (unsigned short*)alloc(TOK * 192 * 2);
  float*          cbufA = (float*)alloc(TOK * 192 * 4);
  float*          cbufB = (float*)alloc(TOK * 24 * 4);
  unsigned short* t1    = (unsigned short*)alloc(TOK * 24 * 2);
  unsigned short* t2    = (unsigned short*)alloc(TOK * 24 * 2);
  float*          psum  = (float*)alloc(192 * 128 * 4);
  float*          psq   = (float*)alloc(192 * 128 * 4);
  float*          stats = (float*)alloc(192 * 2 * 4);

  // ---- weight / bias prep ----
  k_cvt_pad<<<CEILDIV(576 * 192, 256), 256, 0, stream>>>(qkv_w, wqkv, 576, 192, 576);
  k_cvt_pad<<<CEILDIV(192 * 192, 256), 256, 0, stream>>>(proj_w, wproj, 192, 192, 192);
  k_cvt_pad<<<CEILDIV(192 * 768, 256), 256, 0, stream>>>(fc_w, wfc, 192, 768, 192);
  k_cvt_pad<<<CEILDIV(192 * 192, 256), 256, 0, stream>>>(b1_w, wb1, 192, 192, 192);
  k_cvt_pad<<<CEILDIV(32 * 192, 256), 256, 0, stream>>>(b3a_w, wb3a, 24, 192, 32);
  k_cvt_pad<<<CEILDIV(32 * 192, 256), 256, 0, stream>>>(b5a_w, wb5a, 24, 192, 32);
  k_cvt_pad<<<CEILDIV(192 * 192, 256), 256, 0, stream>>>(bp_w, wbp, 192, 192, 192);
  k_prep_convw<<<CEILDIV(27 * 192 * 32, 256), 256, 0, stream>>>(b3b_w, wb3b, 192, 192);
  k_prep_convw<<<CEILDIV(27 * 32 * 32, 256), 256, 0, stream>>>(b5b_w, wb5b, 24, 32);
  k_prep_convw<<<CEILDIV(27 * 192 * 32, 256), 256, 0, stream>>>(b5c_w, wb5c, 192, 192);
  k_prep_bias<<<CEILDIV(6 * 4096, 256), 256, 0, stream>>>(rpb, bias6);

  // ---- fused windowed attention (139264 B dynamic LDS per WG) ----
  const int ATTN_SMEM = 64 * 192 * 2 + 64 * 576 * 2 + 64 * 64 * 4 + 64 * 192 * 2;
  (void)hipFuncSetAttribute((const void*)k_attn,
                            hipFuncAttributeMaxDynamicSharedMemorySize, 160 * 1024);
  k_attn<<<1024, 256, ATTN_SMEM, stream>>>(x, g1, beta1, wqkv, qkv_b, bias6,
                                           wproj, proj_b, xattn);

  // ---- LN2 ----
  k_ln<<<1024, 256, 0, stream>>>(xattn, g2, beta2, hbuf);

  // ---- branch 1: 1x1 conv 192->192, BN, GELU -> cat[:,0:192] ----
  k_gemm<<<1024, 256, 0, stream>>>(hbuf, wb1, 192, 12, b1_b, cbufA, 192, 192);
  k_stat1<<<dim3(192, 128), 256, 0, stream>>>(cbufA, 192, psum, psq);
  k_stat2<<<192, 128, 0, stream>>>(psum, psq, stats);
  k_bngelu<<<49152, 256, 0, stream>>>(cbufA, 192, stats, b1_g, b1_be, catb, 768, 0);

  // ---- branch 3: 1x1 192->24, then 3^3 24->192 -> cat[:,192:384] ----
  k_gemm<<<1024, 256, 0, stream>>>(hbuf, wb3a, 192, 2, b3a_b, cbufB, 24, 24);
  k_stat1<<<dim3(24, 128), 256, 0, stream>>>(cbufB, 24, psum, psq);
  k_stat2<<<24, 128, 0, stream>>>(psum, psq, stats);
  k_bngelu<<<6144, 256, 0, stream>>>(cbufB, 24, stats, b3a_g, b3a_be, t1, 24, 0);
  k_conv3<<<1024, 256, 0, stream>>>(t1, wb3b, b3b_b, cbufA, 192, 192, 192);
  k_stat1<<<dim3(192, 128), 256, 0, stream>>>(cbufA, 192, psum, psq);
  k_stat2<<<192, 128, 0, stream>>>(psum, psq, stats);
  k_bngelu<<<49152, 256, 0, stream>>>(cbufA, 192, stats, b3b_g, b3b_be, catb, 768, 192);

  // ---- branch 5: 1x1 192->24, 3^3 24->24, 3^3 24->192 -> cat[:,384:576] ----
  k_gemm<<<1024, 256, 0, stream>>>(hbuf, wb5a, 192, 2, b5a_b, cbufB, 24, 24);
  k_stat1<<<dim3(24, 128), 256, 0, stream>>>(cbufB, 24, psum, psq);
  k_stat2<<<24, 128, 0, stream>>>(psum, psq, stats);
  k_bngelu<<<6144, 256, 0, stream>>>(cbufB, 24, stats, b5a_g, b5a_be, t1, 24, 0);
  k_conv3<<<1024, 256, 0, stream>>>(t1, wb5b, b5b_b, cbufB, 24, 24, 32);
  k_stat1<<<dim3(24, 128), 256, 0, stream>>>(cbufB, 24, psum, psq);
  k_stat2<<<24, 128, 0, stream>>>(psum, psq, stats);
  k_bngelu<<<6144, 256, 0, stream>>>(cbufB, 24, stats, b5b_g, b5b_be, t2, 24, 0);
  k_conv3<<<1024, 256, 0, stream>>>(t2, wb5c, b5c_b, cbufA, 192, 192, 192);
  k_stat1<<<dim3(192, 128), 256, 0, stream>>>(cbufA, 192, psum, psq);
  k_stat2<<<192, 128, 0, stream>>>(psum, psq, stats);
  k_bngelu<<<49152, 256, 0, stream>>>(cbufA, 192, stats, b5c_g, b5c_be, catb, 768, 384);

  // ---- pool branch: avgpool 3^3 then 1x1 192->192 -> cat[:,576:768] ----
  k_pool<<<1024, 256, 0, stream>>>(hbuf, pooled);
  k_gemm<<<1024, 256, 0, stream>>>(pooled, wbp, 192, 12, bp_b, cbufA, 192, 192);
  k_stat1<<<dim3(192, 128), 256, 0, stream>>>(cbufA, 192, psum, psq);
  k_stat2<<<192, 128, 0, stream>>>(psum, psq, stats);
  k_bngelu<<<49152, 256, 0, stream>>>(cbufA, 192, stats, bp_g, bp_be, catb, 768, 576);

  // ---- fc + residual -> d_out ----
  k_fc<<<1024, 256, 0, stream>>>(catb, wfc, fc_b, xattn, out);
}
